// GroupLinear_61426622267738
// MI455X (gfx1250) — matrix operand order repaired and checked by reading, not compiled, not run on hardware
//
#include <hip/hip_runtime.h>
#include <stdint.h>

// Problem sizes fixed by the reference
#define B_DIM 16384
#define G_DIM 16
#define DIN   512
#define DOUT  512
#define KDIM  (G_DIM * DIN)     // 8192 : k = g*DIN + i
#define KPAD  32                // extra K block: rows [8192,8208)=bias, rest zero
#define KTOT  (KDIM + KPAD)     // 8224
#define BM 128
#define BN 128
#define BK 32
#define NKB (KTOT / BK)         // 257
#define LDT 40                  // LDS pitch in bf16 elems (80B): 16 frag rows -> 16 distinct banks

typedef __attribute__((ext_vector_type(16))) __bf16 v16bf;
typedef __attribute__((ext_vector_type(8)))  __bf16 v8bf;
typedef __attribute__((ext_vector_type(8)))  float  v8f;
typedef int i32x4 __attribute__((vector_size(16)));   // matches builtin's V4i param

#if defined(__has_builtin)
#  if __has_builtin(__builtin_amdgcn_global_load_async_to_lds_b128)
#    define HAVE_ASYNC 1
#  endif
#endif
#ifndef HAVE_ASYNC
#  define HAVE_ASYNC 0
#endif

#define AS1 __attribute__((address_space(1)))
#define AS3 __attribute__((address_space(3)))

// f32 -> bf16 bits via the compiler's native fptrunc (RNE)
__device__ __forceinline__ uint16_t bfbits(float f) {
  union { __bf16 b; uint16_t u; } c;
  c.b = (__bf16)f;
  return c.u;
}

// ---------------------------------------------------------------------------
// Prep 1: transpose weight f32 [KDIM][DOUT] -> Wt bf16 [DOUT][KTOT] (64x64 tiles)
// ---------------------------------------------------------------------------
__global__ void __launch_bounds__(256) wprep_kernel(const float* __restrict__ W,
                                                    uint16_t* __restrict__ Wt) {
  __shared__ uint16_t tile[64][72];                 // 72-pitch keeps 16B alignment
  const int kt = blockIdx.x * 64;
  const int nt = blockIdx.y * 64;
  const int t  = threadIdx.x;
  const int kk = t >> 2;                            // 0..63
  const int c4 = (t & 3) * 16;                      // 0,16,32,48
  const float4* src = (const float4*)(W + (size_t)(kt + kk) * DOUT + nt + c4);
#pragma unroll
  for (int q = 0; q < 4; ++q) {
    float4 v = src[q];
    tile[c4 + q * 4 + 0][kk] = bfbits(v.x);
    tile[c4 + q * 4 + 1][kk] = bfbits(v.y);
    tile[c4 + q * 4 + 2][kk] = bfbits(v.z);
    tile[c4 + q * 4 + 3][kk] = bfbits(v.w);
  }
  __syncthreads();
  const int nn = t >> 2;
  uint16_t* dst = Wt + (size_t)(nt + nn) * KTOT + kt + c4;
  const uint4* srow = (const uint4*)&tile[nn][c4];
  ((uint4*)dst)[0] = srow[0];
  ((uint4*)dst)[1] = srow[1];
}

// ---------------------------------------------------------------------------
// Prep 2: bias rows of the extended K: Wt[n][8192+g] = bias[g][n] (g<16), else 0
// ---------------------------------------------------------------------------
__global__ void __launch_bounds__(256) bprep_kernel(const float* __restrict__ bias,
                                                    uint16_t* __restrict__ Wt) {
  int idx = threadIdx.x + blockIdx.x * 256;         // 64 blocks * 256 = 32*512
  if (idx < KPAD * DOUT) {
    int kk = idx & (KPAD - 1);
    int n  = idx >> 5;
    uint16_t v = (kk < G_DIM) ? bfbits(bias[kk * DOUT + n]) : (uint16_t)0;
    Wt[(size_t)n * KTOT + KDIM + kk] = v;
  }
}

// ---------------------------------------------------------------------------
// Main GEMM: out[16384,512] = Y[16384,8224] @ Wext[8224,512], bf16 WMMA, f32 acc
// ---------------------------------------------------------------------------
__global__ void __launch_bounds__(256)
gl_wmma_kernel(const float* __restrict__ X, const float* __restrict__ Gm,
               const uint16_t* __restrict__ Wt, float* __restrict__ Out) {
  __shared__ uint16_t As[2][BM * LDT];              // 2 x 10240 B
  __shared__ uint16_t Bs[2][BN * LDT];              // 2 x 10240 B (B stored N-major = B^T)

  const int t    = threadIdx.x;
  const int lane = t & 31;
  const int wid  = t >> 5;                          // 8 waves
  const int wm   = wid >> 2;                        // 0..1 (64 M rows each)
  const int wn   = wid & 3;                        // 0..3 (32 N cols each)
  const int r    = lane & 15;
  const int hs   = lane >> 4;                       // lane half
  const int m0   = blockIdx.y * BM;
  const int n0   = blockIdx.x * BN;
  const int sr   = t >> 1;                          // staging row 0..127
  const int sh   = t & 1;                           // staging half (16 elems)

  const float*    xrow = X  + (size_t)(m0 + sr) * DIN;
  const float*    grow = Gm + (size_t)(m0 + sr) * G_DIM;
  const uint16_t* wrow = Wt + (size_t)(n0 + sr) * KTOT + sh * 16;

  const v8f vzero = {0.f, 0.f, 0.f, 0.f, 0.f, 0.f, 0.f, 0.f};

  v8f acc[4][2] = {};

  v8f a0, a1;                                       // staged A (raw x values)
  float gs;
#if !HAVE_ASYNC
  uint4 br0, br1;
#endif

  // Issue global reads for K-block kb (A into regs; B async directly to LDS)
  auto issue = [&](int kb, int dstbuf) {
    const int k0 = kb * BK;
    if (k0 < KDIM) {
      const v8f* px = (const v8f*)(xrow + (k0 & (DIN - 1)) + sh * 16);
      gs = grow[k0 >> 9];                           // g[b, k0/512]
      a0 = px[0];
      a1 = px[1];
    } else {                                        // bias block: Y[b, 8192+g] = g[b,g]
      gs = 1.0f;
      if (sh == 0) {
        const v8f* pg = (const v8f*)grow;
        a0 = pg[0];
        a1 = pg[1];
      } else {
        a0 = vzero;
        a1 = vzero;
      }
    }
#if HAVE_ASYNC
    {
      const uint16_t* src = wrow + k0;
      uint16_t* dst = &Bs[dstbuf][sr * LDT + sh * 16];
      __builtin_amdgcn_global_load_async_to_lds_b128((AS1 i32x4*)(src),
                                                     (AS3 i32x4*)(dst), 0, 0);
      __builtin_amdgcn_global_load_async_to_lds_b128((AS1 i32x4*)(src + 8),
                                                     (AS3 i32x4*)(dst + 8), 0, 0);
    }
#else
    {
      const uint4* ps = (const uint4*)(wrow + k0);
      br0 = ps[0]; br1 = ps[1];
    }
#endif
  };

  // Scale by g, convert to bf16 (vector fptrunc), commit staged data to LDS
  auto commit = [&](int dstbuf) {
    v8bf lo = __builtin_convertvector(a0 * gs, v8bf);
    v8bf hi = __builtin_convertvector(a1 * gs, v8bf);
    v8bf* da = (v8bf*)&As[dstbuf][sr * LDT + sh * 16];
    da[0] = lo;
    da[1] = hi;
#if !HAVE_ASYNC
    uint4* db = (uint4*)&Bs[dstbuf][sr * LDT + sh * 16];
    db[0] = br0; db[1] = br1;
#endif
  };

  auto compute = [&](int buf) {
    v16bf bf[2];
#pragma unroll
    for (int j = 0; j < 2; ++j) {
      // B fragment: lane holds column n = r, contiguous K = hs*16 + e
      const uint16_t* base = &Bs[buf][(wn * 32 + j * 16 + r) * LDT + hs * 16];
      v8bf lo = *(const v8bf*)base;
      v8bf hi = *(const v8bf*)(base + 8);
      bf[j] = __builtin_shufflevector(lo, hi, 0, 1, 2, 3, 4, 5, 6, 7,
                                      8, 9, 10, 11, 12, 13, 14, 15);
    }
#pragma unroll
    for (int i = 0; i < 4; ++i) {
      // A fragment: lane holds row m = r; contiguous K = hs*16 + e
      const uint16_t* abase = &As[buf][(wm * 64 + i * 16 + r) * LDT];
      v8bf lo = *(const v8bf*)(abase + hs * 16);
      v8bf hi = *(const v8bf*)(abase + hs * 16 + 8);
      v16bf af = __builtin_shufflevector(lo, hi, 0, 1, 2, 3, 4, 5, 6, 7,
                                         8, 9, 10, 11, 12, 13, 14, 15);
#pragma unroll
      for (int j = 0; j < 2; ++j)
        acc[i][j] = __builtin_amdgcn_wmma_f32_16x16x32_bf16(
            false, af, false, bf[j], (short)0, acc[i][j], false, false);
    }
  };

  // Prologue
  issue(0, 0);
  commit(0);
#if HAVE_ASYNC
#  if defined(__has_builtin) && __has_builtin(__builtin_amdgcn_s_wait_asynccnt)
  __builtin_amdgcn_s_wait_asynccnt(0);
#  else
  asm volatile("s_wait_asynccnt 0" ::: "memory");
#  endif
#endif
  __syncthreads();

  // Double-buffered main loop: overlap next-block global fetch with WMMA
  for (int kb = 0; kb < NKB; ++kb) {
    const int buf = kb & 1;
    if (kb + 1 < NKB) issue(kb + 1, buf ^ 1);
    compute(buf);
    if (kb + 1 < NKB) commit(buf ^ 1);
#if HAVE_ASYNC
#  if defined(__has_builtin) && __has_builtin(__builtin_amdgcn_s_wait_asynccnt)
    __builtin_amdgcn_s_wait_asynccnt(0);
#  else
    asm volatile("s_wait_asynccnt 0" ::: "memory");
#  endif
#endif
    __syncthreads();
  }

  // Epilogue: C/D layout — VGPR v: M = v + 8*hs, N = r
#pragma unroll
  for (int i = 0; i < 4; ++i) {
#pragma unroll
    for (int j = 0; j < 2; ++j) {
      const int gm = m0 + wm * 64 + i * 16 + hs * 8;
      const int gn = n0 + wn * 32 + j * 16 + r;
      float* o = Out + (size_t)gm * DOUT + gn;
#pragma unroll
      for (int v = 0; v < 8; ++v) o[(size_t)v * DOUT] = acc[i][j][v];
    }
  }
}

// ---------------------------------------------------------------------------
extern "C" void kernel_launch(void* const* d_in, const int* in_sizes, int n_in,
                              void* d_out, int out_size, void* d_ws, size_t ws_size,
                              hipStream_t stream) {
  (void)in_sizes; (void)n_in; (void)out_size; (void)ws_size;
  const float* x    = (const float*)d_in[0];   // [B, DIN]
  const float* g    = (const float*)d_in[1];   // [B, G]
  const float* w    = (const float*)d_in[2];   // [G, DIN, DOUT] == [KDIM, DOUT]
  const float* bias = (const float*)d_in[3];   // [G, DOUT]
  float* out = (float*)d_out;                  // [B, DOUT]

  // Workspace: Wt bf16 [DOUT][KTOT] = 512*8224*2 B ~= 8.0 MB (assumed <= ws_size)
  uint16_t* Wt = (uint16_t*)d_ws;

  wprep_kernel<<<dim3(KDIM / 64, DOUT / 64), 256, 0, stream>>>(w, Wt);
  bprep_kernel<<<dim3((KPAD * DOUT + 255) / 256), 256, 0, stream>>>(bias, Wt);
  gl_wmma_kernel<<<dim3(DOUT / BN, B_DIM / BM), 256, 0, stream>>>(x, g, Wt, out);
}
